// PointNet2Backbone_9105330667530
// MI455X (gfx1250) — matrix operand order repaired and checked by reading, not compiled, not run on hardware
//
#include <hip/hip_runtime.h>
#include <math.h>

typedef __attribute__((ext_vector_type(16))) _Float16 v16h;
typedef __attribute__((ext_vector_type(8)))  _Float16 v8h;
typedef __attribute__((ext_vector_type(8)))  float    v8f;

// ---------------- problem constants (from reference) ----------------
#define NPTS   16384
#define NC1    3276          // int(0.2*N)
#define NC2    819           // int(0.25*N1)
#define KNB    64
#define R1SQ   (0.05f*0.05f)
#define R2SQ   (0.1f*0.1f)
#define NC1P   3280          // padded to /16
#define NC2P   832
#define MROW1  (NC1*KNB)     // 209664 rows, /16 ok
#define MROW2  (NC2*KNB)     // 52416 rows, /16 ok

// ==================== utility kernels ====================
__global__ void fill_h_kernel(_Float16* p, long n) {
  long i = (long)blockIdx.x * blockDim.x + threadIdx.x;
  if (i < n) p[i] = (_Float16)0.f;
}

__global__ void cvt_h_kernel(const float* __restrict__ s, _Float16* __restrict__ d, long n) {
  long i = (long)blockIdx.x * blockDim.x + threadIdx.x;
  if (i < n) d[i] = (_Float16)s[i];
}

// weight [fi,fo] f32 -> TRANSPOSED [fo, fiPad] f16, zero-padded along K.
// Transposed layout makes each B fragment (K = half*16 .. +15 of one output col)
// a contiguous 32-byte run -> two global_load_b128 per fragment.
__global__ void cvt_wT_kernel(const float* __restrict__ s, int fi, int fo,
                              _Float16* __restrict__ d, int fiPad) {
  int i = blockIdx.x * blockDim.x + threadIdx.x;
  if (i >= fo * fiPad) return;
  int n = i / fiPad, k = i - n * fiPad;
  d[i] = (k < fi) ? (_Float16)s[k * fo + n] : (_Float16)0.f;
}

__global__ void gatherpos_kernel(const float* __restrict__ pos, const int* __restrict__ idx,
                                 int m, float* __restrict__ out) {
  int i = blockIdx.x * blockDim.x + threadIdx.x;
  if (i >= m * 3) return;
  int r = i / 3, c = i - r * 3;
  out[i] = pos[idx[r] * 3 + c];
}

__global__ void seti_kernel(int* p, int v) {
  if (blockIdx.x == 0 && threadIdx.x == 0) *p = v;
}

__global__ void minlab_kernel(const int* __restrict__ label, int n, int* out) {
  int i = blockIdx.x * blockDim.x + threadIdx.x;
  if (i < n) atomicMin(out, label[i]);
}

// ==================== farthest point sampling ====================
__global__ void fps_kernel(const float* __restrict__ pos, int n, int m,
                           int* __restrict__ out, float* __restrict__ mind) {
  const int T = blockDim.x;
  int t = threadIdx.x, lane = t & 31, wid = t >> 5;
  __shared__ float s_last[3];
  __shared__ float s_wval[32];
  __shared__ int   s_widx[32];
  for (int i = t; i < n; i += T) mind[i] = 3.4e38f;
  if (t == 0) out[0] = 0;
  if (t < 3) s_last[t] = pos[t];
  __syncthreads();
  for (int it = 1; it < m; ++it) {
    float lx = s_last[0], ly = s_last[1], lz = s_last[2];
    float bv = -1.f; int bi = 0;
    for (int i = t; i < n; i += T) {
      float dx = pos[i*3+0]-lx, dy = pos[i*3+1]-ly, dz = pos[i*3+2]-lz;
      float d  = dx*dx + dy*dy + dz*dz;
      float md = mind[i]; md = d < md ? d : md; mind[i] = md;
      if (md > bv) { bv = md; bi = i; }
    }
    #pragma unroll
    for (int off = 16; off > 0; off >>= 1) {
      float ov = __shfl_down(bv, off, 32);
      int   oi = __shfl_down(bi, off, 32);
      if (ov > bv) { bv = ov; bi = oi; }
    }
    if (lane == 0) { s_wval[wid] = bv; s_widx[wid] = bi; }
    __syncthreads();
    if (t == 0) {
      float best = s_wval[0]; int besti = s_widx[0];
      int nw = T >> 5;
      for (int w = 1; w < nw; ++w)
        if (s_wval[w] > best) { best = s_wval[w]; besti = s_widx[w]; }
      out[it] = besti;
      s_last[0] = pos[besti*3+0]; s_last[1] = pos[besti*3+1]; s_last[2] = pos[besti*3+2];
    }
    __syncthreads();
  }
}

// ==================== ball query (first-K within radius, one wave per center) ====================
__global__ void ball_query_kernel(const float* __restrict__ pos, int n,
                                  const float* __restrict__ centers, int m, float r2,
                                  int* __restrict__ idx, int* __restrict__ cnt) {
  int wave = (int)((blockIdx.x * blockDim.x + threadIdx.x) >> 5);
  int lane = threadIdx.x & 31;
  if (wave >= m) return;                      // uniform per wave
  float cx = centers[wave*3+0], cy = centers[wave*3+1], cz = centers[wave*3+2];
  int count = 0;
  for (int base = 0; base < n; base += 32) {
    int p = base + lane;
    bool ok = false;
    if (p < n) {
      float dx = pos[p*3+0]-cx, dy = pos[p*3+1]-cy, dz = pos[p*3+2]-cz;
      ok = (dx*dx + dy*dy + dz*dz) <= r2;
    }
    unsigned mask = (unsigned)__ballot(ok);   // wave32: low 32 bits
    int pre = __popc(mask & ((1u << lane) - 1u));
    if (ok && (count + pre) < KNB) idx[(size_t)wave*KNB + count + pre] = p;
    count += __popc(mask);
    if (count >= KNB) break;
  }
  if (count > KNB) count = KNB;
  if (lane == 0) cnt[wave] = count;
}

// ==================== grouped-feature build ====================
__global__ void group_feats_kernel(const _Float16* __restrict__ feat, int featC, int ldf,
                                   const float* __restrict__ pos,
                                   const float* __restrict__ centers,
                                   const int* __restrict__ nidx, const int* __restrict__ cnt,
                                   int m, _Float16* __restrict__ A, int lda) {
  int row = blockIdx.x * blockDim.x + threadIdx.x;
  if (row >= m * KNB) return;
  int c = row / KNB, slot = row - c * KNB;
  bool valid = slot < cnt[c];
  _Float16* dst = A + (size_t)row * lda;
  if (valid) {
    int p = nidx[(size_t)c*KNB + slot];
    const _Float16* f = feat + (size_t)p * ldf;
    for (int j = 0; j < featC; ++j) dst[j] = f[j];
    for (int j = 0; j < 3; ++j) dst[featC+j] = (_Float16)(pos[p*3+j] - centers[c*3+j]);
  } else {
    for (int j = 0; j < featC + 3; ++j) dst[j] = (_Float16)0.f;
  }
}

// ==================== WMMA GEMM ====================
// relu(A[MxK] * W^T + bias).  A row-major f16 (lda mult of 8), Bt transposed
// weights [N x Kpad] f16.  One wave owns a 16x64 output strip: one A fragment
// feeds 4 v_wmma_f32_16x16x32_f16 per 32-wide k-step.  Fragments are loaded as
// b128 pairs straight from memory (each lane's K = half*16..+15 is a contiguous run in
// this storage scheme) -- no scalar gathers.
__device__ __forceinline__ v16h frag_pair(const _Float16* p0, const _Float16* p1) {
  v8h lo = *(const v8h*)p0;
  v8h hi = *(const v8h*)p1;
  return __builtin_shufflevector(lo, hi, 0,1,2,3,4,5,6,7,8,9,10,11,12,13,14,15);
}

__global__ void gemm_bias_relu_wmma(const _Float16* __restrict__ A, int lda,
                                    const _Float16* __restrict__ Bt, int ldbt,
                                    const float* __restrict__ bias,
                                    _Float16* __restrict__ Ch, float* __restrict__ Cf,
                                    int ldc, int M, int K) {
  int wave = threadIdx.x >> 5;
  int lane = threadIdx.x & 31;
  int mTile = blockIdx.x * (blockDim.x >> 5) + wave;
  int m0 = mTile * 16;
  int n0 = blockIdx.y * 64;
  if (m0 >= M) return;                         // uniform per wave -> EXEC stays full
  int half = lane >> 4, r = lane & 15;
  const _Float16* aRow = A + (size_t)(m0 + r) * lda;
  v8f acc0 = {}, acc1 = {}, acc2 = {}, acc3 = {};
  for (int k0 = 0; k0 < K; k0 += 32) {
    if (k0 + 32 < K)                           // global_prefetch_b8 next A tile
      __builtin_prefetch(aRow + k0 + 32 + half*8, 0, 1);
    // A fragment: rows m0+r, K = half*16..+15 (contiguous)
    v16h a = frag_pair(aRow + k0 + half*16, aRow + k0 + half*16 + 8);
    // B fragments: weight rows n0+16t+r, K = half*16..+15 (contiguous, transposed store)
    const _Float16* bBase = Bt + (size_t)(n0 + r) * ldbt + k0 + half*16;
    v16h b0 = frag_pair(bBase,                 bBase + 8);
    v16h b1 = frag_pair(bBase + 16*(size_t)ldbt,     bBase + 16*(size_t)ldbt + 8);
    v16h b2 = frag_pair(bBase + 32*(size_t)ldbt,     bBase + 32*(size_t)ldbt + 8);
    v16h b3 = frag_pair(bBase + 48*(size_t)ldbt,     bBase + 48*(size_t)ldbt + 8);
    acc0 = __builtin_amdgcn_wmma_f32_16x16x32_f16(false, a, false, b0, (short)0, acc0, false, false);
    acc1 = __builtin_amdgcn_wmma_f32_16x16x32_f16(false, a, false, b1, (short)0, acc1, false, false);
    acc2 = __builtin_amdgcn_wmma_f32_16x16x32_f16(false, a, false, b2, (short)0, acc2, false, false);
    acc3 = __builtin_amdgcn_wmma_f32_16x16x32_f16(false, a, false, b3, (short)0, acc3, false, false);
  }
  #pragma unroll
  for (int t = 0; t < 4; ++t) {
    v8f acc = (t == 0) ? acc0 : (t == 1) ? acc1 : (t == 2) ? acc2 : acc3;
    int nn = n0 + t*16 + r;
    float bv = bias[nn];
    #pragma unroll
    for (int i = 0; i < 8; ++i) {
      int mm = m0 + i + half*8;
      float v = acc[i] + bv;
      v = v > 0.f ? v : 0.f;                   // relu on every layer (reference)
      Ch[(size_t)mm * ldc + nn] = (_Float16)v;
      if (Cf) Cf[(size_t)mm * ldc + nn] = v;
    }
  }
}

// ==================== max pool over valid neighbors ====================
__global__ void maxpool_kernel(const _Float16* __restrict__ H, int ldh,
                               const int* __restrict__ cnt, int m, int C,
                               _Float16* __restrict__ out, int ldo) {
  int i = blockIdx.x * blockDim.x + threadIdx.x;
  if (i >= m * C) return;
  int c = i / C, ch = i - c * C;
  int nv = cnt[c]; if (nv < 1) nv = 1;
  float best = -3.4e38f;
  const _Float16* base = H + (size_t)c * KNB * ldh + ch;
  for (int s = 0; s < nv; ++s) {
    float v = (float)base[(size_t)s * ldh];
    best = v > best ? v : best;
  }
  out[(size_t)c * ldo + ch] = (_Float16)best;
}

__global__ void colmax_kernel(const _Float16* __restrict__ H, int ldh, int m, int C,
                              _Float16* __restrict__ out) {
  int ch = blockIdx.x * blockDim.x + threadIdx.x;
  if (ch >= C) return;
  float best = -3.4e38f;
  for (int i = 0; i < m; ++i) {
    float v = (float)H[(size_t)i * ldh + ch];
    best = v > best ? v : best;
  }
  out[ch] = (_Float16)best;
}

// ==================== concat helpers ====================
__global__ void copycols_h_kernel(const _Float16* __restrict__ src, int lds_, int m, int C,
                                  _Float16* __restrict__ dst, int ldd, int colOff) {
  int i = blockIdx.x * blockDim.x + threadIdx.x;
  if (i >= m * C) return;
  int r = i / C, c = i - r * C;
  dst[(size_t)r * ldd + colOff + c] = src[(size_t)r * lds_ + c];
}

__global__ void copycols_f_kernel(const float* __restrict__ src, int lds_, int m, int C,
                                  _Float16* __restrict__ dst, int ldd, int colOff) {
  int i = blockIdx.x * blockDim.x + threadIdx.x;
  if (i >= m * C) return;
  int r = i / C, c = i - r * C;
  dst[(size_t)r * ldd + colOff + c] = (_Float16)src[(size_t)r * lds_ + c];
}

__global__ void bcastrow_kernel(const _Float16* __restrict__ row, int C, int m,
                                _Float16* __restrict__ dst, int ldd, int colOff) {
  int i = blockIdx.x * blockDim.x + threadIdx.x;
  if (i >= m * C) return;
  int r = i / C, c = i - r * C;
  dst[(size_t)r * ldd + colOff + c] = row[c];
}

// ==================== 3-NN inverse-distance interpolation ====================
__global__ void knn3_kernel(const float* __restrict__ posDst, int mDst,
                            const float* __restrict__ posSrc, int mSrc,
                            const _Float16* __restrict__ featSrc, int ldf, int C,
                            _Float16* __restrict__ out, int ldo, int colOff) {
  int i = blockIdx.x * blockDim.x + threadIdx.x;
  if (i >= mDst) return;
  float px = posDst[i*3+0], py = posDst[i*3+1], pz = posDst[i*3+2];
  float d0 = 3.4e38f, d1 = 3.4e38f, d2 = 3.4e38f;
  int   i0 = 0, i1 = 0, i2 = 0;
  for (int j = 0; j < mSrc; ++j) {
    float dx = posSrc[j*3+0]-px, dy = posSrc[j*3+1]-py, dz = posSrc[j*3+2]-pz;
    float d = dx*dx + dy*dy + dz*dz;
    if (d < d0)      { d2=d1; i2=i1; d1=d0; i1=i0; d0=d; i0=j; }
    else if (d < d1) { d2=d1; i2=i1; d1=d;  i1=j; }
    else if (d < d2) { d2=d;  i2=j; }
  }
  float w0 = 1.f/fmaxf(d0,1e-16f), w1 = 1.f/fmaxf(d1,1e-16f), w2 = 1.f/fmaxf(d2,1e-16f);
  float ws = w0 + w1 + w2; w0 /= ws; w1 /= ws; w2 /= ws;
  const _Float16* f0 = featSrc + (size_t)i0 * ldf;
  const _Float16* f1 = featSrc + (size_t)i1 * ldf;
  const _Float16* f2 = featSrc + (size_t)i2 * ldf;
  _Float16* d = out + (size_t)i * ldo + colOff;
  for (int c = 0; c < C; ++c)
    d[c] = (_Float16)(w0*(float)f0[c] + w1*(float)f1[c] + w2*(float)f2[c]);
}

// ==================== ArcFace head ====================
__global__ void arcface_kernel(const float* __restrict__ u0, const float* __restrict__ arcW,
                               const int* __restrict__ label, const int* __restrict__ minlab,
                               int n, float* __restrict__ out) {
  int i = blockIdx.x * blockDim.x + threadIdx.x;
  if (i >= n) return;
  const float* u = u0 + (size_t)i * 128;
  float nx = 0.f;
  for (int c = 0; c < 128; ++c) nx += u[c]*u[c];
  float inx = rsqrtf(fmaxf(nx, 1e-24f));
  int lab = label[i] - ((*minlab >= 1) ? 1 : 0);
  for (int cls = 0; cls < 3; ++cls) {
    const float* w = arcW + cls * 128;
    float nw = 0.f, dot = 0.f;
    for (int c = 0; c < 128; ++c) { nw += w[c]*w[c]; dot += u[c]*w[c]; }
    float cosv = dot * inx * rsqrtf(fmaxf(nw, 1e-24f));
    cosv = fminf(fmaxf(cosv, -1.f + 1e-7f), 1.f - 1e-7f);
    float tgt = cosf(acosf(cosv) + 0.5f);
    out[(size_t)i*3 + cls] = 30.f * ((cls == lab) ? tgt : cosv);
  }
}

// ==================== host orchestration ====================
extern "C" void kernel_launch(void* const* d_in, const int* in_sizes, int n_in,
                              void* d_out, int out_size, void* d_ws, size_t ws_size,
                              hipStream_t stream) {
  const float* x     = (const float*)d_in[0];
  const float* pos   = (const float*)d_in[1];
  const int*   label = (const int*)d_in[2];
  const float* arcW  = (const float*)d_in[3];
  const float *fp1W0=(const float*)d_in[4],  *fp1b0=(const float*)d_in[5];
  const float *fp1W1=(const float*)d_in[6],  *fp1b1=(const float*)d_in[7];
  const float *fp1W2=(const float*)d_in[8],  *fp1b2=(const float*)d_in[9];
  const float *fp2W0=(const float*)d_in[10], *fp2b0=(const float*)d_in[11];
  const float *fp2W1=(const float*)d_in[12], *fp2b1=(const float*)d_in[13];
  const float *fp3W0=(const float*)d_in[14], *fp3b0=(const float*)d_in[15];
  const float *fp3W1=(const float*)d_in[16], *fp3b1=(const float*)d_in[17];
  const float *sa1W0=(const float*)d_in[18], *sa1b0=(const float*)d_in[19];
  const float *sa1W1=(const float*)d_in[20], *sa1b1=(const float*)d_in[21];
  const float *sa1W2=(const float*)d_in[22], *sa1b2=(const float*)d_in[23];
  const float *sa2W0=(const float*)d_in[24], *sa2b0=(const float*)d_in[25];
  const float *sa2W1=(const float*)d_in[26], *sa2b1=(const float*)d_in[27];
  const float *sa2W2=(const float*)d_in[28], *sa2b2=(const float*)d_in[29];
  const float *sa3W0=(const float*)d_in[30], *sa3b0=(const float*)d_in[31];
  const float *sa3W1=(const float*)d_in[32], *sa3b1=(const float*)d_in[33];
  const float *sa3W2=(const float*)d_in[34], *sa3b2=(const float*)d_in[35];

  char* wsBase = (char*)d_ws;
  size_t wsOff = 0;
  auto alloc = [&](size_t bytes) -> void* {
    wsOff = (wsOff + 255) & ~(size_t)255;
    void* p = wsBase + wsOff; wsOff += bytes; return p;
  };
  float*    mind  = (float*)alloc(4ull*NPTS);
  int*  fps1_idx  = (int*)  alloc(4ull*NC1);
  float*    pos1  = (float*)alloc(4ull*NC1*3);
  int*   bq1_idx  = (int*)  alloc(4ull*NC1*KNB);
  int*      cnt1  = (int*)  alloc(4ull*NC1);
  _Float16* xh    = (_Float16*)alloc(2ull*NPTS*3);
  _Float16* A1    = (_Float16*)alloc(2ull*MROW1*32);
  _Float16* H1a   = (_Float16*)alloc(2ull*MROW1*64);
  _Float16* H1b   = (_Float16*)alloc(2ull*MROW1*64);
  _Float16* H1c   = (_Float16*)alloc(2ull*MROW1*128);
  _Float16* x1    = (_Float16*)alloc(2ull*NC1P*128);
  int*  fps2_idx  = (int*)  alloc(4ull*NC2);
  float*    pos2  = (float*)alloc(4ull*NC2*3);
  int*   bq2_idx  = (int*)  alloc(4ull*NC2*KNB);
  int*      cnt2  = (int*)  alloc(4ull*NC2);
  _Float16* A2    = (_Float16*)alloc(2ull*MROW2*160);
  _Float16* H2a   = (_Float16*)alloc(2ull*MROW2*128);
  _Float16* H2b   = (_Float16*)alloc(2ull*MROW2*128);
  _Float16* H2c   = (_Float16*)alloc(2ull*MROW2*256);
  _Float16* x2    = (_Float16*)alloc(2ull*NC2P*256);
  _Float16* A3    = (_Float16*)alloc(2ull*NC2P*288);
  _Float16* H3a   = (_Float16*)alloc(2ull*NC2P*256);
  _Float16* H3b   = (_Float16*)alloc(2ull*NC2P*512);
  _Float16* H3c   = (_Float16*)alloc(2ull*NC2P*1024);
  _Float16* x3    = (_Float16*)alloc(2ull*1024);
  _Float16* A4    = (_Float16*)alloc(2ull*NC2P*1280);
  _Float16* H4a   = (_Float16*)alloc(2ull*NC2P*256);
  _Float16* u2    = (_Float16*)alloc(2ull*NC2P*256);
  _Float16* A5    = (_Float16*)alloc(2ull*NC1P*384);
  _Float16* H5a   = (_Float16*)alloc(2ull*NC1P*256);
  _Float16* u1    = (_Float16*)alloc(2ull*NC1P*128);
  _Float16* A6    = (_Float16*)alloc(2ull*NPTS*160);
  _Float16* H6a   = (_Float16*)alloc(2ull*NPTS*128);
  _Float16* H6b   = (_Float16*)alloc(2ull*NPTS*128);
  _Float16* H6c   = (_Float16*)alloc(2ull*NPTS*128);
  float*    u0    = (float*)   alloc(4ull*NPTS*128);
  int*    minlab  = (int*)     alloc(4);
  _Float16* sa1w0 = (_Float16*)alloc(2ull*32*64);
  _Float16* sa1w1 = (_Float16*)alloc(2ull*64*64);
  _Float16* sa1w2 = (_Float16*)alloc(2ull*64*128);
  _Float16* sa2w0 = (_Float16*)alloc(2ull*160*128);
  _Float16* sa2w1 = (_Float16*)alloc(2ull*128*128);
  _Float16* sa2w2 = (_Float16*)alloc(2ull*128*256);
  _Float16* sa3w0 = (_Float16*)alloc(2ull*288*256);
  _Float16* sa3w1 = (_Float16*)alloc(2ull*256*512);
  _Float16* sa3w2 = (_Float16*)alloc(2ull*512*1024);
  _Float16* fp3w0 = (_Float16*)alloc(2ull*1280*256);
  _Float16* fp3w1 = (_Float16*)alloc(2ull*256*256);
  _Float16* fp2w0 = (_Float16*)alloc(2ull*384*256);
  _Float16* fp2w1 = (_Float16*)alloc(2ull*256*128);
  _Float16* fp1w0 = (_Float16*)alloc(2ull*160*128);
  _Float16* fp1w1 = (_Float16*)alloc(2ull*128*128);
  _Float16* fp1w2 = (_Float16*)alloc(2ull*128*128);

  auto fillh = [&](_Float16* p, long n) {
    fill_h_kernel<<<dim3((unsigned)((n+255)/256)), dim3(256), 0, stream>>>(p, n);
  };
  auto cvtW = [&](const float* s, int fi, int fo, _Float16* d, int fiPad) {
    int n = fo * fiPad;
    cvt_wT_kernel<<<dim3((n+255)/256), dim3(256), 0, stream>>>(s, fi, fo, d, fiPad);
  };
  auto gemm = [&](const _Float16* A, int lda, const _Float16* Bt, int ldbt, const float* bias,
                  _Float16* Ch, float* Cf, int ldc, int M, int Nout, int K) {
    dim3 grd((M/16 + 7)/8, Nout/64);
    gemm_bias_relu_wmma<<<grd, dim3(256), 0, stream>>>(A, lda, Bt, ldbt, bias, Ch, Cf, ldc, M, K);
  };

  // ---- zero-fill padded GEMM inputs / outputs reused as inputs ----
  fillh(A1, (long)MROW1*32);  fillh(A2, (long)MROW2*160);
  fillh(A3, (long)NC2P*288);  fillh(A4, (long)NC2P*1280);
  fillh(A5, (long)NC1P*384);  fillh(A6, (long)NPTS*160);
  fillh(x1, (long)NC1P*128);  fillh(x2, (long)NC2P*256);

  // ---- weights f32 -> transposed f16 [fo, fiPad] ----
  cvtW(sa1W0,   6,   64, sa1w0,   32);
  cvtW(sa1W1,  64,   64, sa1w1,   64);
  cvtW(sa1W2,  64,  128, sa1w2,   64);
  cvtW(sa2W0, 131,  128, sa2w0,  160);
  cvtW(sa2W1, 128,  128, sa2w1,  128);
  cvtW(sa2W2, 128,  256, sa2w2,  128);
  cvtW(sa3W0, 259,  256, sa3w0,  288);
  cvtW(sa3W1, 256,  512, sa3w1,  256);
  cvtW(sa3W2, 512, 1024, sa3w2,  512);
  cvtW(fp3W0,1280,  256, fp3w0, 1280);
  cvtW(fp3W1, 256,  256, fp3w1,  256);
  cvtW(fp2W0, 384,  256, fp2w0,  384);
  cvtW(fp2W1, 256,  128, fp2w1,  256);
  cvtW(fp1W0, 131,  128, fp1w0,  160);
  cvtW(fp1W1, 128,  128, fp1w1,  128);
  cvtW(fp1W2, 128,  128, fp1w2,  128);

  cvt_h_kernel<<<dim3((NPTS*3+255)/256), dim3(256), 0, stream>>>(x, xh, (long)NPTS*3);

  // ---- SA1 ----
  fps_kernel<<<1, 1024, 0, stream>>>(pos, NPTS, NC1, fps1_idx, mind);
  gatherpos_kernel<<<dim3((NC1*3+255)/256), dim3(256), 0, stream>>>(pos, fps1_idx, NC1, pos1);
  ball_query_kernel<<<dim3((NC1*32+255)/256), dim3(256), 0, stream>>>(pos, NPTS, pos1, NC1,
                                                                      R1SQ, bq1_idx, cnt1);
  group_feats_kernel<<<dim3((MROW1+255)/256), dim3(256), 0, stream>>>(xh, 3, 3, pos, pos1,
                                                                      bq1_idx, cnt1, NC1, A1, 32);
  gemm(A1,  32, sa1w0, 32,  sa1b0, H1a, nullptr, 64,  MROW1,  64, 32);
  gemm(H1a, 64, sa1w1, 64,  sa1b1, H1b, nullptr, 64,  MROW1,  64, 64);
  gemm(H1b, 64, sa1w2, 64,  sa1b2, H1c, nullptr, 128, MROW1, 128, 64);
  maxpool_kernel<<<dim3((NC1*128+255)/256), dim3(256), 0, stream>>>(H1c, 128, cnt1, NC1, 128, x1, 128);

  // ---- SA2 ----
  fps_kernel<<<1, 1024, 0, stream>>>(pos1, NC1, NC2, fps2_idx, mind);
  gatherpos_kernel<<<dim3((NC2*3+255)/256), dim3(256), 0, stream>>>(pos1, fps2_idx, NC2, pos2);
  ball_query_kernel<<<dim3((NC2*32+255)/256), dim3(256), 0, stream>>>(pos1, NC1, pos2, NC2,
                                                                      R2SQ, bq2_idx, cnt2);
  group_feats_kernel<<<dim3((MROW2+255)/256), dim3(256), 0, stream>>>(x1, 128, 128, pos1, pos2,
                                                                      bq2_idx, cnt2, NC2, A2, 160);
  gemm(A2,  160, sa2w0, 160, sa2b0, H2a, nullptr, 128, MROW2, 128, 160);
  gemm(H2a, 128, sa2w1, 128, sa2b1, H2b, nullptr, 128, MROW2, 128, 128);
  gemm(H2b, 128, sa2w2, 128, sa2b2, H2c, nullptr, 256, MROW2, 256, 128);
  maxpool_kernel<<<dim3((NC2*256+255)/256), dim3(256), 0, stream>>>(H2c, 256, cnt2, NC2, 256, x2, 256);

  // ---- SA3 (global) ----
  copycols_h_kernel<<<dim3((NC2*256+255)/256), dim3(256), 0, stream>>>(x2, 256, NC2, 256, A3, 288, 0);
  copycols_f_kernel<<<dim3((NC2*3+255)/256),   dim3(256), 0, stream>>>(pos2, 3, NC2, 3,   A3, 288, 256);
  gemm(A3,  288, sa3w0, 288, sa3b0, H3a, nullptr, 256,  NC2P,  256, 288);
  gemm(H3a, 256, sa3w1, 256, sa3b1, H3b, nullptr, 512,  NC2P,  512, 256);
  gemm(H3b, 512, sa3w2, 512, sa3b2, H3c, nullptr, 1024, NC2P, 1024, 512);
  colmax_kernel<<<dim3((1024+255)/256), dim3(256), 0, stream>>>(H3c, 1024, NC2, 1024, x3);

  // ---- FP3 ----
  bcastrow_kernel<<<dim3((NC2*1024+255)/256), dim3(256), 0, stream>>>(x3, 1024, NC2, A4, 1280, 0);
  copycols_h_kernel<<<dim3((NC2*256+255)/256), dim3(256), 0, stream>>>(x2, 256, NC2, 256, A4, 1280, 1024);
  gemm(A4,  1280, fp3w0, 1280, fp3b0, H4a, nullptr, 256, NC2P, 256, 1280);
  gemm(H4a,  256, fp3w1,  256, fp3b1, u2,  nullptr, 256, NC2P, 256, 256);

  // ---- FP2 ----
  knn3_kernel<<<dim3((NC1+255)/256), dim3(256), 0, stream>>>(pos1, NC1, pos2, NC2,
                                                             u2, 256, 256, A5, 384, 0);
  copycols_h_kernel<<<dim3((NC1*128+255)/256), dim3(256), 0, stream>>>(x1, 128, NC1, 128, A5, 384, 256);
  gemm(A5,  384, fp2w0, 384, fp2b0, H5a, nullptr, 256, NC1P, 256, 384);
  gemm(H5a, 256, fp2w1, 256, fp2b1, u1,  nullptr, 128, NC1P, 128, 256);

  // ---- FP1 ----
  knn3_kernel<<<dim3((NPTS+255)/256), dim3(256), 0, stream>>>(pos, NPTS, pos1, NC1,
                                                              u1, 128, 128, A6, 160, 0);
  copycols_h_kernel<<<dim3((NPTS*3+255)/256), dim3(256), 0, stream>>>(xh, 3, NPTS, 3, A6, 160, 128);
  gemm(A6,  160, fp1w0, 160, fp1b0, H6a, nullptr, 128, NPTS, 128, 160);
  gemm(H6a, 128, fp1w1, 128, fp1b1, H6b, nullptr, 128, NPTS, 128, 128);
  gemm(H6b, 128, fp1w2, 128, fp1b2, H6c, u0,      128, NPTS, 128, 128);

  // ---- ArcFace head ----
  seti_kernel<<<1, 32, 0, stream>>>(minlab, 0x7fffffff);
  minlab_kernel<<<dim3((NPTS+255)/256), dim3(256), 0, stream>>>(label, NPTS, minlab);
  arcface_kernel<<<dim3((NPTS+255)/256), dim3(256), 0, stream>>>(u0, arcW, label, minlab,
                                                                 NPTS, (float*)d_out);
}
